// W4Linear_20306605375842
// MI455X (gfx1250) — compile-verified, matrix operand order checked by reading, not run on hardware
//
#include <hip/hip_runtime.h>

// CDNA5 / gfx1250 W4A16 quantized GEMM using v_wmma_f32_16x16x32_f16 and the
// Tensor Data Mover (tensor_load_to_lds) for double-buffered x-tile staging.
// out[M, OC] = x[M, IC] @ dequant(qweight)[OC, IC]^T + bias
// M = 8192, IC = 4096, OC = 11008, group size 128 (32 groups).
//
// Block tile: 128(M) x 128(N), 8 waves; each wave owns a distinct 16-col N
// tile so every dequantized B fragment feeds 8 WMMA M-tiles (no duplicated
// weight fetch/dequant within a block).

typedef __attribute__((ext_vector_type(16))) _Float16 v16h;
typedef __attribute__((ext_vector_type(8)))  _Float16 v8h;
typedef __attribute__((ext_vector_type(2)))  _Float16 h2;
typedef __attribute__((ext_vector_type(8)))  float    v8f;
typedef __attribute__((ext_vector_type(4)))  int      v4i;
typedef __attribute__((ext_vector_type(4)))  unsigned int u32x4;
typedef __attribute__((ext_vector_type(8)))  int      i32x8;
typedef __attribute__((ext_vector_type(4)))  int      i32x4;

#define IC 4096
#define OC 11008
#define GS 128
#define NG 32
#define LDS_STRIDE 136                     // 128 + 8 halves pad per row
#define TILE_HALVES (128 * LDS_STRIDE)
#define TILE_BYTES  (TILE_HALVES * 2)

union FragH {
    v16h v;
    v8h  h[2];
    h2   p[8];
};

union PackF { unsigned u; h2 h; };

// Dequantize two uint4 codes (each in an int32) to fp16 in one packed FMA:
// f16bits(0x6400 | q) == 1024 + q   ->   w = (1024+q)*s + (z - 1024*s)
__device__ __forceinline__ h2 dq2(int qa, int qb, h2 s2, h2 z2) {
    PackF p;
    p.u = ((unsigned)qa | ((unsigned)qb << 16)) | 0x64006400u;
    return p.h * s2 + z2;   // v_pk_fma_f16
}

__device__ __forceinline__ void dequantFrag(FragH& B, const int* __restrict__ qrow,
                                            int kGlob, int kOff, h2 s2, h2 z2) {
    v4i q0 = *(const v4i*)(qrow + kGlob + kOff);
    v4i q1 = *(const v4i*)(qrow + kGlob + kOff + 4);
    v4i q2 = *(const v4i*)(qrow + kGlob + 16 + kOff);
    v4i q3 = *(const v4i*)(qrow + kGlob + 16 + kOff + 4);
    B.p[0] = dq2(q0.x, q0.y, s2, z2);
    B.p[1] = dq2(q0.z, q0.w, s2, z2);
    B.p[2] = dq2(q1.x, q1.y, s2, z2);
    B.p[3] = dq2(q1.z, q1.w, s2, z2);
    B.p[4] = dq2(q2.x, q2.y, s2, z2);
    B.p[5] = dq2(q2.z, q2.w, s2, z2);
    B.p[6] = dq2(q3.x, q3.y, s2, z2);
    B.p[7] = dq2(q3.z, q3.w, s2, z2);
}

#if __has_builtin(__builtin_amdgcn_tensor_load_to_lds)
#define USE_TDM 1
// TDM DMA: load a 128x128 fp16 tile (row stride IC) into LDS, inserting a
// 4-dword pad after every 64 dwords (=> 272B = 136-half LDS row stride).
__device__ __forceinline__ void tdm_load_tile(unsigned ldsByteOff,
                                              const _Float16* __restrict__ gsrc) {
    unsigned long long ga = (unsigned long long)(uintptr_t)gsrc;
    u32x4 g0;
    g0.x = 0x1u;                                              // count=1, user D#
    g0.y = ldsByteOff;                                        // lds_addr (bytes)
    g0.z = (unsigned)(ga & 0xFFFFFFFFu);                      // global_addr[31:0]
    g0.w = (unsigned)((ga >> 32) & 0x01FFFFFFu) | (2u << 30); // addr[56:32] | type=2

    i32x8 g1;
    g1[0] = (int)((1u << 16)     // data_size = 2 bytes
                | (1u << 20)     // pad_enable
                | (5u << 22)     // pad_interval = 64 dwords (256B tile row)
                | (3u << 25));   // pad_amount   = 4 dwords (16B pad)
    g1[1] = (int)(((unsigned)IC & 0xFFFFu) << 16);   // tensor_dim0[15:0] = 4096
    g1[2] = 0;                                        // tensor_dim0[31:16]=0, td1[15:0]=0
    g1[3] = (int)(0x10u | (128u << 16));              // td1[31:16]=0x10 (1M rows), tile_dim0=128
    g1[4] = 128;                                      // tile_dim1=128, tile_dim2=0
    g1[5] = IC;                                       // tensor_dim0_stride = 4096 elements
    g1[6] = 0;
    g1[7] = 0;

    i32x4 zz4 = {0, 0, 0, 0};                         // 2D tensor: groups 2/3 unused
    i32x8 zz8 = {0, 0, 0, 0, 0, 0, 0, 0};
    __builtin_amdgcn_tensor_load_to_lds(g0, g1, zz4, zz4, zz8, 0);
}
#else
#define USE_TDM 0
#endif

__global__ __launch_bounds__(256)
void w4a16_wmma_kernel(const _Float16* __restrict__ x,
                       const int*      __restrict__ qweight,
                       const _Float16* __restrict__ scales,
                       const _Float16* __restrict__ zeros,
                       const _Float16* __restrict__ bias,
                       _Float16*       __restrict__ out)
{
    extern __shared__ _Float16 ldsA[];   // 2 * TILE_HALVES halves (double buffer)

    const int tid   = threadIdx.x;
    const int lane  = tid & 31;
    const int wave  = tid >> 5;              // 0..7 -> distinct 16-col N tile
    const int l16   = lane & 15;
    const int kOff  = (lane < 16) ? 0 : 8;   // 16-bit A/B fragment K sub-layout

    const int mBlock = blockIdx.y * 128;
    const int nBlock = blockIdx.x * 128;

    // Output column handled by this lane (one N tile per wave)
    const int n0 = nBlock + wave * 16 + l16;
    const int* qrow = qweight + (size_t)n0 * IC;

    const _Float16* xBase = x + (size_t)mBlock * IC;

    v8f acc[8] = {};

#if USE_TDM
    const unsigned ldsBase = (unsigned)(uintptr_t)(void*)ldsA;
    if (wave == 0) tdm_load_tile(ldsBase, xBase);           // prologue: tile g=0
#endif

    for (int g = 0; g < NG; ++g) {
#if USE_TDM
        if (wave == 0) __builtin_amdgcn_s_wait_tensorcnt(0);  // tile g landed
        __syncthreads();                                      // visible to all waves
        if (wave == 0 && g + 1 < NG)                          // overlap next DMA
            tdm_load_tile(ldsBase + ((g + 1) & 1) * TILE_BYTES,
                          xBase + (size_t)(g + 1) * GS);
        const _Float16* tile = ldsA + (g & 1) * TILE_HALVES;
#else
        // Manual staging fallback: x[mBlock:+128, g*128:+128] -> LDS buffer 0
        {
            const _Float16* xg = xBase + (size_t)g * GS;
            int row = tid >> 3;
            const int col = (tid & 7) * 16;
            #pragma unroll
            for (int p = 0; p < 4; ++p) {
                const v8h* src = (const v8h*)(xg + (size_t)row * IC + col);
                v8h d0 = src[0];
                v8h d1 = src[1];
                *(v8h*)&ldsA[row * LDS_STRIDE + col]     = d0;
                *(v8h*)&ldsA[row * LDS_STRIDE + col + 8] = d1;
                row += 32;
            }
        }
        __syncthreads();
        const _Float16* tile = ldsA;
#endif

        // Per-(group, column) dequant constants (magic-offset form)
        const _Float16 s0 = scales[g * OC + n0];
        const _Float16 za0 = zeros[g * OC + n0] - s0 * (_Float16)1024.0f;
        const h2 s2 = {s0, s0}, z2 = {za0, za0};

        // Prefetch next group's weight row while we compute on this one
        if (g + 1 < NG) {
            __builtin_prefetch(qrow + (g + 1) * GS + kOff, 0, 1);
        }

        #pragma unroll
        for (int kk = 0; kk < 4; ++kk) {
            const int kb = kk * 32;   // K offset within the staged group tile

            // ---- One B fragment (32x16 f16): packed magic dequant ----
            FragH B;
            dequantFrag(B, qrow, g * GS + kb, kOff, s2, z2);

            // ---- 8 WMMA M-tiles share the B fragment ----
            #pragma unroll
            for (int mi = 0; mi < 8; ++mi) {
                const int aRow = mi * 16 + l16;
                FragH A;
                A.h[0] = *(const v8h*)&tile[aRow * LDS_STRIDE + kb + kOff];
                A.h[1] = *(const v8h*)&tile[aRow * LDS_STRIDE + kb + 16 + kOff];
                acc[mi] = __builtin_amdgcn_wmma_f32_16x16x32_f16(
                    false, A.v, false, B.v, (short)0, acc[mi], false, false);
            }
        }

#if !USE_TDM
        __syncthreads();   // release LDS buffer before next staging pass
#endif
    }

    // ---- Epilogue: bias add + fp16 store per 16x16 C layout ----
    const float bf = (float)bias[n0];
    const int mTop = mBlock + ((lane < 16) ? 0 : 8);

    #pragma unroll
    for (int mi = 0; mi < 8; ++mi) {
        const int mBase = mTop + mi * 16;
        #pragma unroll
        for (int i = 0; i < 8; ++i) {
            const size_t row = (size_t)(mBase + i);
            out[row * OC + n0] = (_Float16)(acc[mi][i] + bf);
        }
    }
}

extern "C" void kernel_launch(void* const* d_in, const int* in_sizes, int n_in,
                              void* d_out, int out_size, void* d_ws, size_t ws_size,
                              hipStream_t stream) {
    (void)n_in; (void)out_size; (void)d_ws; (void)ws_size;
    const _Float16* x   = (const _Float16*)d_in[0];
    const int*      qw  = (const int*)d_in[1];
    const _Float16* sc  = (const _Float16*)d_in[2];
    const _Float16* zp  = (const _Float16*)d_in[3];
    const _Float16* bs  = (const _Float16*)d_in[4];
    _Float16*       out = (_Float16*)d_out;

    const int M = in_sizes[0] / IC;          // 4*2048 = 8192
    dim3 grid(OC / 128, M / 128);            // 86 x 64 blocks
    const size_t ldsBytes = 2 * TILE_BYTES;  // double-buffered x tile (69632 B)
    w4a16_wmma_kernel<<<grid, 256, ldsBytes, stream>>>(x, qw, sc, zp, bs, out);
}
